// CleanMultiHeadAttention_17154099380426
// MI455X (gfx1250) — matrix operand order repaired and checked by reading, not compiled, not run on hardware
//
#include <hip/hip_runtime.h>
#include <hip/hip_bf16.h>

typedef __bf16 bf16;
typedef __attribute__((ext_vector_type(16))) __bf16 v16bf;
typedef __attribute__((ext_vector_type(8)))  __bf16 v8bf;
typedef __attribute__((ext_vector_type(8)))  float  v8f;
typedef __attribute__((ext_vector_type(4)))  float  v4f;

#define DIM 2048
#define NHEADS 16
#define HDIM 128
#define NB 2
#define SEQ 2048

__device__ __forceinline__ v8f zero8() {
  v8f z = {0.f, 0.f, 0.f, 0.f, 0.f, 0.f, 0.f, 0.f};
  return z;
}

// ---------------------------------------------------------------------------
// CDNA5 async copy: global -> LDS, 16 bytes per lane, tracked by ASYNCcnt.
__device__ __forceinline__ void async_copy_b128(void* lds_dst, const void* gsrc) {
  unsigned l = (unsigned)(uintptr_t)lds_dst;  // low 32 bits = LDS offset
  asm volatile("global_load_async_to_lds_b128 %0, %1, off"
               :
               : "v"(l), "v"(gsrc)
               : "memory");
}
__device__ __forceinline__ void wait_asynccnt0() {
  asm volatile("s_wait_asynccnt 0x0" ::: "memory");
}

// ---------------------------------------------------------------------------
// WMMA fragment loaders (wave32, ISA 7.12.2 layouts). Contiguous 16B halves ->
// the compiler merges each into two ds_load_b128 / global_load_b128.
// A fragment: 16x32 bf16 (MxK), source row-major [m][k].
__device__ __forceinline__ v16bf load_frag_a(const bf16* p, int lane, int stride) {
  const int row  = lane & 15;
  const int koff = (lane >= 16) ? 8 : 0;
  const bf16* r = p + (size_t)row * stride;
  v16bf f;
#pragma unroll
  for (int v = 0; v < 8; ++v) {
    int k = koff + (v >> 2) * 16 + (v & 3) * 2;
    f[2 * v]     = r[k];
    f[2 * v + 1] = r[k + 1];
  }
  return f;
}

// B fragment: 32x16 bf16 (KxN). Source stored as [n][k] row-major, B[k][n]=p[n*stride+k].
__device__ __forceinline__ v16bf load_frag_b(const bf16* p, int lane, int stride) {
  const int n    = lane & 15;
  const int koff = (lane >= 16) ? 8 : 0;
  const bf16* r = p + (size_t)n * stride;
  v16bf f;
#pragma unroll
  for (int v = 0; v < 8; ++v) {
    int k = koff + (v >> 2) * 16 + (v & 3) * 2;
    f[2 * v]     = r[k];
    f[2 * v + 1] = r[k + 1];
  }
  return f;
}

// Pack 8 f32 -> 8 bf16 (v_cvt_pk_bf16_f32 pairs).
__device__ __forceinline__ v8bf pack8(v4f f0, v4f f1) {
  v8bf pk;
#pragma unroll
  for (int e = 0; e < 4; ++e) {
    pk[e]     = (bf16)f0[e];
    pk[4 + e] = (bf16)f1[e];
  }
  return pk;
}

// ---------------------------------------------------------------------------
// Kernel 1: y = x @ W^T for one head (128 cols) x 64 rows, fused RoPE.
// mode: 1 = RoPE + head-major [b][h][l][hd] store (Q,K)
//       2 = no RoPE + transposed [b][h][hd][l] store (V, for the P@V B operand)
// grid: (NB*SEQ/64, NHEADS), block 256 (8 waves).
__global__ __launch_bounds__(256) void qkv_rope_kernel(
    const float* __restrict__ x, const float* __restrict__ w,
    const float* __restrict__ rope, bf16* __restrict__ out, int mode) {
  constexpr int BM = 64, BK = 32, LDAB = BK + 8;
  __shared__ __align__(16) bf16  Alds[BM * LDAB];
  __shared__ __align__(16) bf16  Blds[HDIM * LDAB];
  __shared__ __align__(16) float Olds[BM * HDIM];

  const int tid  = threadIdx.x;
  const int wave = tid >> 5, lane = tid & 31;
  const int row0 = blockIdx.x * BM;  // token-row block in [0, NB*SEQ)
  const int h    = blockIdx.y;
  const int col0 = h * HDIM;

  // Fixed per-thread staging slices (fully unrolled, b128 traffic only).
  const int am  = tid >> 2, ac0 = (tid & 3) * 8;   // A: 8 f32 -> 8 bf16
  const int bn  = tid >> 1, bc0 = (tid & 1) * 16;  // B: 16 f32 -> 16 bf16

  v8f acc[4];
#pragma unroll
  for (int m = 0; m < 4; ++m) acc[m] = zero8();

  for (int k0 = 0; k0 < DIM; k0 += BK) {
    {
      const float* src = &x[(size_t)(row0 + am) * DIM + k0 + ac0];
      v4f f0 = *(const v4f*)(src);
      v4f f1 = *(const v4f*)(src + 4);
      *(v8bf*)&Alds[am * LDAB + ac0] = pack8(f0, f1);
    }
    {
      const float* src = &w[(size_t)(col0 + bn) * DIM + k0 + bc0];
#pragma unroll
      for (int half = 0; half < 2; ++half) {
        v4f f0 = *(const v4f*)(src + half * 8);
        v4f f1 = *(const v4f*)(src + half * 8 + 4);
        *(v8bf*)&Blds[bn * LDAB + bc0 + half * 8] = pack8(f0, f1);
      }
    }
    if (k0 + BK < DIM) {
      __builtin_prefetch(&x[(size_t)(row0 + am) * DIM + k0 + BK + ac0], 0, 1);
      __builtin_prefetch(&w[(size_t)(col0 + bn) * DIM + k0 + BK + bc0], 0, 1);
    }
    __syncthreads();
    v16bf bfr = load_frag_b(&Blds[(wave * 16) * LDAB], lane, LDAB);
#pragma unroll
    for (int m = 0; m < 4; ++m) {
      v16bf af = load_frag_a(&Alds[(m * 16) * LDAB], lane, LDAB);
      acc[m] = __builtin_amdgcn_wmma_f32_16x16x32_bf16(false, af, false, bfr,
                                                       (short)0, acc[m], false, false);
    }
    __syncthreads();
  }

  // Spill accumulators so the RoPE epilogue can pair hd with hd +/- 64.
  const int ncol = wave * 16 + (lane & 15);
  const int rb   = (lane >= 16) ? 8 : 0;
#pragma unroll
  for (int m = 0; m < 4; ++m)
#pragma unroll
    for (int r = 0; r < 8; ++r)
      Olds[(m * 16 + rb + r) * HDIM + ncol] = acc[m][r];
  __syncthreads();

  if (mode == 1) {  // RoPE, head-major store (coalesced along hd)
#pragma unroll
    for (int it = 0; it < BM * HDIM / 256; ++it) {
      int i = it * 256 + tid;
      int m = i >> 7, hd = i & 127;
      int grow = row0 + m;
      int b = grow / SEQ, l = grow % SEQ;
      float v = Olds[m * HDIM + hd];
      float ang = rope[(size_t)l * HDIM + hd];
      float partner =
          (hd < 64) ? -Olds[m * HDIM + hd + 64] : Olds[m * HDIM + hd - 64];
      float o = v * __cosf(ang) + partner * __sinf(ang);
      out[(((size_t)b * NHEADS + h) * SEQ + l) * HDIM + hd] = (bf16)o;
    }
  } else {  // V: transposed [b][h][hd][l] store (coalesced along l)
#pragma unroll
    for (int it = 0; it < BM * HDIM / 256; ++it) {
      int i = it * 256 + tid;
      int hd = i >> 6, m = i & 63;
      int grow = row0 + m;
      int b = grow / SEQ, l = grow % SEQ;
      out[(((size_t)b * NHEADS + h) * HDIM + hd) * SEQ + l] =
          (bf16)Olds[m * HDIM + hd];
    }
  }
}

// ---------------------------------------------------------------------------
// Kernel 2: flash attention for one (b, h, 64-query block), 64-key chunks.
// K/V^T chunks staged with GLOBAL_LOAD_ASYNC_TO_LDS_B128 (ASYNCcnt).
// grid: (SEQ/64, NHEADS, NB), block 256 (8 waves).
__global__ __launch_bounds__(256) void flash_attn_kernel(
    const bf16* __restrict__ Q, const bf16* __restrict__ K,
    const bf16* __restrict__ VT, bf16* __restrict__ O) {
  constexpr int BM = 64, CHUNK = 64;
  constexpr int LDK   = HDIM + 8;   // K tile  [kk][d]  bf16 stride (272 B)
  constexpr int LDV   = CHUNK + 8;  // V^T tile [d][kk] bf16 stride (144 B)
  constexpr int LDS_S = CHUNK + 4;  // S f32 stride 68 (272 B) -> P bf16 stride 136
  __shared__ __align__(16) bf16  Klds[CHUNK * LDK];
  __shared__ __align__(16) bf16  VTlds[HDIM * LDV];
  __shared__ __align__(16) float Sbuf[BM * LDS_S];
  __shared__ float m_run[BM], l_run[BM], alpha_s[BM];

  const int tid  = threadIdx.x;
  const int wave = tid >> 5, lane = tid & 31;
  const int q0   = blockIdx.x * BM;
  const int h    = blockIdx.y, b = blockIdx.z;

  const size_t headoff = ((size_t)b * NHEADS + h) * SEQ * HDIM;
  const bf16* Qh  = Q + headoff;
  const bf16* Kh  = K + headoff;
  const bf16* VTh = VT + headoff;  // [hd][l]

  // S-gemm split: wave owns key n-tile nw and query m-tiles {m0, m0+1}.
  const int nw = wave & 3;
  const int m0 = (wave >> 2) * 2;
  v16bf qf[2][4];  // Q fragments live in registers for the whole key loop
#pragma unroll
  for (int mi = 0; mi < 2; ++mi)
#pragma unroll
    for (int ks = 0; ks < 4; ++ks)
      qf[mi][ks] =
          load_frag_a(Qh + (size_t)(q0 + (m0 + mi) * 16) * HDIM + ks * 32, lane, HDIM);

  if (tid < BM) {
    m_run[tid] = -3.0e38f;
    l_run[tid] = 0.f;
  }

  v8f acc_o[4];  // wave owns d n-tile = wave, m-tiles 0..3
#pragma unroll
  for (int m = 0; m < 4; ++m) acc_o[m] = zero8();

  const float scale = 0.0883883476483184f;  // 1/sqrt(HDIM)
  const int rb = (lane >= 16) ? 8 : 0;
  const int nc = lane & 15;

  // Fixed per-thread staging slices.
  const int kkk = tid >> 2, kc0 = (tid & 3) * 32;  // K: 32 bf16 = 4 x b128
  const int vd  = tid >> 1, vc0 = (tid & 1) * 32;  // VT: 32 bf16 = 4 x b128

  __syncthreads();

  for (int j = 0; j < SEQ; j += CHUNK) {
    // Async-stage K chunk [kk][d] and V^T chunk [d][kk].
    {
      const bf16* src = &Kh[(size_t)(j + kkk) * HDIM + kc0];
      bf16* dst = &Klds[kkk * LDK + kc0];
#pragma unroll
      for (int q = 0; q < 4; ++q) async_copy_b128(dst + q * 8, src + q * 8);
    }
    {
      const bf16* src = &VTh[(size_t)vd * SEQ + j + vc0];
      bf16* dst = &VTlds[vd * LDV + vc0];
#pragma unroll
      for (int q = 0; q < 4; ++q) async_copy_b128(dst + q * 8, src + q * 8);
    }
    wait_asynccnt0();
    __syncthreads();

    // S = (Q K^T) * scale
#pragma unroll
    for (int mi = 0; mi < 2; ++mi) {
      v8f s = zero8();
#pragma unroll
      for (int ks = 0; ks < 4; ++ks) {
        v16bf kf = load_frag_b(&Klds[(nw * 16) * LDK + ks * 32], lane, LDK);
        s = __builtin_amdgcn_wmma_f32_16x16x32_bf16(false, qf[mi][ks], false, kf,
                                                    (short)0, s, false, false);
      }
#pragma unroll
      for (int r = 0; r < 8; ++r)
        Sbuf[((m0 + mi) * 16 + rb + r) * LDS_S + nw * 16 + nc] = s[r] * scale;
    }
    __syncthreads();

    // Online softmax: one thread per query row; P overwrites S in place as bf16
    // (byte 2c of a row always lands inside an already-consumed float).
    if (tid < BM) {
      float* srow  = &Sbuf[tid * LDS_S];
      float  mprev = m_run[tid];
      float  mx    = mprev;
      for (int c = 0; c < CHUNK; ++c) mx = fmaxf(mx, srow[c]);
      float al   = __expf(mprev - mx);
      bf16* prow = (bf16*)srow;
      float sum  = 0.f;
      for (int c = 0; c < CHUNK; ++c) {
        float p = __expf(srow[c] - mx);
        prow[c] = (bf16)p;
        sum += p;
      }
      m_run[tid]   = mx;
      alpha_s[tid] = al;
      l_run[tid]   = l_run[tid] * al + sum;
    }
    __syncthreads();

    // acc_o = acc_o * alpha + P @ V   (V fragments depend only on ks: hoisted)
    v16bf vf[2];
#pragma unroll
    for (int ks = 0; ks < 2; ++ks)
      vf[ks] = load_frag_b(&VTlds[(wave * 16) * LDV + ks * 32], lane, LDV);
#pragma unroll
    for (int m = 0; m < 4; ++m) {
      v8f pv = zero8();
#pragma unroll
      for (int ks = 0; ks < 2; ++ks) {
        v16bf pf = load_frag_a((const bf16*)&Sbuf[(m * 16) * LDS_S] + ks * 32,
                               lane, 2 * LDS_S);
        pv = __builtin_amdgcn_wmma_f32_16x16x32_bf16(false, pf, false, vf[ks],
                                                     (short)0, pv, false, false);
      }
#pragma unroll
      for (int r = 0; r < 8; ++r)
        acc_o[m][r] = acc_o[m][r] * alpha_s[m * 16 + rb + r] + pv[r];
    }
    __syncthreads();
  }

  // Normalize and store O (bf16, token-major [B*L][DIM]).
#pragma unroll
  for (int m = 0; m < 4; ++m)
#pragma unroll
    for (int r = 0; r < 8; ++r) {
      int row = m * 16 + rb + r;
      int d   = wave * 16 + nc;
      float o = acc_o[m][r] / l_run[row];
      O[((size_t)(b * SEQ + q0 + row)) * DIM + h * HDIM + d] = (bf16)o;
    }
}

// ---------------------------------------------------------------------------
// Kernel 3: out = O @ Wo^T + bo (f32 output). A staged via async LDS copies.
// grid: (NB*SEQ/64, DIM/128), block 256.
__global__ __launch_bounds__(256) void out_proj_kernel(
    const bf16* __restrict__ A, const float* __restrict__ wo,
    const float* __restrict__ bo, float* __restrict__ out) {
  constexpr int BM = 64, BN = 128, BK = 32, LDAB = BK + 8;
  __shared__ __align__(16) bf16 Alds[BM * LDAB];
  __shared__ __align__(16) bf16 Blds[BN * LDAB];

  const int tid  = threadIdx.x;
  const int wave = tid >> 5, lane = tid & 31;
  const int row0 = blockIdx.x * BM;
  const int col0 = blockIdx.y * BN;

  const int am = tid >> 2, ac0 = (tid & 3) * 8;   // A: one b128 per thread
  const int bn = tid >> 1, bc0 = (tid & 1) * 16;  // B: 16 f32 -> 16 bf16

  v8f acc[4];
#pragma unroll
  for (int m = 0; m < 4; ++m) acc[m] = zero8();

  for (int k0 = 0; k0 < DIM; k0 += BK) {
    async_copy_b128(&Alds[am * LDAB + ac0],
                    &A[(size_t)(row0 + am) * DIM + k0 + ac0]);
    {
      const float* src = &wo[(size_t)(col0 + bn) * DIM + k0 + bc0];
#pragma unroll
      for (int half = 0; half < 2; ++half) {
        v4f f0 = *(const v4f*)(src + half * 8);
        v4f f1 = *(const v4f*)(src + half * 8 + 4);
        *(v8bf*)&Blds[bn * LDAB + bc0 + half * 8] = pack8(f0, f1);
      }
    }
    if (k0 + BK < DIM) {
      __builtin_prefetch(&wo[(size_t)(col0 + bn) * DIM + k0 + BK + bc0], 0, 1);
    }
    wait_asynccnt0();
    __syncthreads();
    v16bf bfr = load_frag_b(&Blds[(wave * 16) * LDAB], lane, LDAB);
#pragma unroll
    for (int m = 0; m < 4; ++m) {
      v16bf af = load_frag_a(&Alds[(m * 16) * LDAB], lane, LDAB);
      acc[m] = __builtin_amdgcn_wmma_f32_16x16x32_bf16(false, af, false, bfr,
                                                       (short)0, acc[m], false, false);
    }
    __syncthreads();
  }

  const int nc  = lane & 15;
  const int rb  = (lane >= 16) ? 8 : 0;
  const int col = col0 + wave * 16 + nc;
  const float bias = bo[col];
#pragma unroll
  for (int m = 0; m < 4; ++m)
#pragma unroll
    for (int r = 0; r < 8; ++r)
      out[(size_t)(row0 + m * 16 + rb + r) * DIM + col] = acc[m][r] + bias;
}

// ---------------------------------------------------------------------------
extern "C" void kernel_launch(void* const* d_in, const int* in_sizes, int n_in,
                              void* d_out, int out_size, void* d_ws, size_t ws_size,
                              hipStream_t stream) {
  (void)in_sizes; (void)n_in; (void)out_size; (void)ws_size;
  const float* x    = (const float*)d_in[0];
  const float* rope = (const float*)d_in[1];
  const float* wq   = (const float*)d_in[2];
  const float* wk   = (const float*)d_in[3];
  const float* wv   = (const float*)d_in[4];
  const float* wo   = (const float*)d_in[5];
  const float* bo   = (const float*)d_in[6];
  float* out = (float*)d_out;

  // Workspace: q,k head-major; v transposed per head; o token-major (4 x 16.8 MB).
  bf16* ws = (bf16*)d_ws;
  const size_t HE = (size_t)NB * NHEADS * SEQ * HDIM;
  bf16* q_ws  = ws;
  bf16* k_ws  = ws + HE;
  bf16* vT_ws = ws + 2 * HE;
  bf16* o_ws  = ws + 3 * HE;

  dim3 blk(256);
  dim3 g1(NB * SEQ / 64, NHEADS);
  qkv_rope_kernel<<<g1, blk, 0, stream>>>(x, wq, rope, q_ws, 1);
  qkv_rope_kernel<<<g1, blk, 0, stream>>>(x, wk, rope, k_ws, 1);
  qkv_rope_kernel<<<g1, blk, 0, stream>>>(x, wv, rope, vT_ws, 2);

  dim3 g2(SEQ / 64, NHEADS, NB);
  flash_attn_kernel<<<g2, blk, 0, stream>>>(q_ws, k_ws, vT_ws, o_ws);

  dim3 g3(NB * SEQ / 64, DIM / 128);
  out_proj_kernel<<<g3, blk, 0, stream>>>(o_ws, wo, bo, out);
}
